// AttnBlock_4982162064074
// MI455X (gfx1250) — matrix operand order repaired and checked by reading, not compiled, not run on hardware
//
#include <hip/hip_runtime.h>
#include <math.h>

// ---------------------------------------------------------------------------
// AttnBlock for MI455X (gfx1250, wave32, WMMA 16x16x32 f16 + TDM async loads).
// Flash-attention structure: S = Q^T K never materialized in HBM.
// K/V tiles are staged to LDS by the Tensor Data Mover (tensor_load_to_lds),
// double-buffered against the WMMA stream via s_wait_tensorcnt.
// ---------------------------------------------------------------------------

typedef __attribute__((ext_vector_type(16))) _Float16 v16h;
typedef __attribute__((ext_vector_type(8)))  _Float16 v8h;
typedef __attribute__((ext_vector_type(8)))  float    v8f;
typedef __attribute__((ext_vector_type(4)))  unsigned v4u;
typedef __attribute__((ext_vector_type(8)))  int      v8i;
typedef __attribute__((ext_vector_type(4)))  int      v4i;

constexpr int kB = 4;
constexpr int kC = 256;
constexpr int kL = 4096;
constexpr float kScale = 0.0625f;      // C^-0.5 = 1/16
constexpr float kEps   = 1e-6f;

static __device__ __forceinline__ v8f wmma_f16(v16h a, v16h b, v8f c) {
  // (neg_a, A, neg_b, B, c_mod, C, reuse_a, reuse_b)
  return __builtin_amdgcn_wmma_f32_16x16x32_f16(false, a, false, b, (short)0, c,
                                                false, false);
}

static __device__ __forceinline__ v16h pack16(v8h lo, v8h hi) {
  v16h r;
#pragma unroll
  for (int t = 0; t < 8; ++t) { r[t] = lo[t]; r[8 + t] = hi[t]; }
  return r;
}

// ---------------------------------------------------------------------------
// TDM: async 2-D f16 tile load Global -> LDS with per-row LDS padding.
//   tile_d0 contiguous elements per row, tile_d1 rows, row stride d0_stride.
//   pad_interval/pad_amount use the D# encodings (interval: 0=2dw..7=256dw,
//   amount: n -> n+1 dwords) so the LDS image gets bank-conflict padding for
//   free. Tracked on TENSORcnt.
// ---------------------------------------------------------------------------
static __device__ __forceinline__ void tdm_load_2d_f16(
    unsigned lds_addr, const _Float16* gaddr, unsigned tensor_d0,
    unsigned tensor_d1, unsigned tile_d0, unsigned tile_d1, unsigned d0_stride,
    unsigned pad_interval, unsigned pad_amount) {
  union { const _Float16* p; unsigned u[2]; } ga;
  ga.p = gaddr;
  v4u g0;
  g0[0] = 1u;                                   // count=1, user descriptor
  g0[1] = lds_addr;                             // lds_addr [63:32]
  g0[2] = ga.u[0];                              // global_addr [95:64]
  g0[3] = (ga.u[1] & 0x01FFFFFFu) | (2u << 30); // global_addr[56:32], type=2
  v8i g1;
  g1[0] = (int)((1u << 16)                      // data_size = 2 bytes
                | (1u << 20)                    // pad_enable
                | (pad_interval << 22) | (pad_amount << 25));
  g1[1] = (int)((tensor_d0 & 0xFFFFu) << 16);   // atomic_barrier=0, td0 lo
  g1[2] = (int)((tensor_d0 >> 16) | ((tensor_d1 & 0xFFFFu) << 16));
  g1[3] = (int)((tensor_d1 >> 16) | (tile_d0 << 16));
  g1[4] = (int)tile_d1;                         // tile_dim1; tile_dim2=0
  g1[5] = (int)d0_stride;                       // tensor_dim0_stride lo
  g1[6] = 0;
  g1[7] = 0;
  v4i z4 = {0, 0, 0, 0};
  v8i z8 = {0, 0, 0, 0, 0, 0, 0, 0};
  __builtin_amdgcn_tensor_load_to_lds(g0, g1, z4, z4, z8, 0);
}

// ---------------------------------------------------------------------------
// 1) GroupNorm statistics: one block per (batch, group); 8 channels x L each.
// ---------------------------------------------------------------------------
__global__ __launch_bounds__(256) void gn_stats(const float* __restrict__ x,
                                                float* __restrict__ mean,
                                                float* __restrict__ rstd) {
  const int g = blockIdx.x;                 // b*32 + group (regions contiguous)
  const float* p = x + (size_t)g * (8 * kL);
  float s = 0.f, ss = 0.f;
  for (int i = threadIdx.x; i < 8 * kL; i += 256) {
    float v = p[i];
    s += v; ss += v * v;
  }
#pragma unroll
  for (int msk = 16; msk >= 1; msk >>= 1) {
    s  += __shfl_xor(s,  msk, 32);
    ss += __shfl_xor(ss, msk, 32);
  }
  __shared__ float sh[16];
  const int wv = threadIdx.x >> 5;
  if ((threadIdx.x & 31) == 0) { sh[wv] = s; sh[8 + wv] = ss; }
  __syncthreads();
  if (threadIdx.x == 0) {
    float S = 0.f, SS = 0.f;
#pragma unroll
    for (int w = 0; w < 8; ++w) { S += sh[w]; SS += sh[8 + w]; }
    const float n = 8.f * kL;
    const float mu = S / n;
    const float var = SS / n - mu * mu;
    mean[g] = mu;
    rstd[g] = rsqrtf(var + kEps);
  }
}

// ---------------------------------------------------------------------------
// 2) Apply GN, emit xn as f16 in [B, L, C] (channel-contiguous) layout.
// ---------------------------------------------------------------------------
__global__ __launch_bounds__(256) void gn_apply(const float* __restrict__ x,
                                                const float* __restrict__ gw,
                                                const float* __restrict__ gb,
                                                const float* __restrict__ mean,
                                                const float* __restrict__ rstd,
                                                _Float16* __restrict__ xnT) {
  const int idx = blockIdx.x * 256 + threadIdx.x;   // over B*C*L = 2^22
  const int b = idx >> 20;
  const int rem = idx & 0xFFFFF;
  const int c = rem >> 12;
  const int l = rem & 0xFFF;
  const int g = (b << 5) | (c >> 3);
  const float v = (x[idx] - mean[g]) * rstd[g] * gw[c] + gb[c];
  xnT[((size_t)b * kL + l) * kC + c] = (_Float16)v;
}

// ---------------------------------------------------------------------------
// 3) fp32 -> f16 weight conversion.
// ---------------------------------------------------------------------------
__global__ __launch_bounds__(256) void cvt16(const float* __restrict__ s,
                                             _Float16* __restrict__ d, int n) {
  const int i = blockIdx.x * 256 + threadIdx.x;
  if (i < n) d[i] = (_Float16)s[i];
}

// ---------------------------------------------------------------------------
// 4) QKV 1x1 conv GEMMs: one wave per 16x16 tile, K=256 in 8 WMMA steps.
//    q,k stored [B,L,C]; v stored [B,C,L].
// ---------------------------------------------------------------------------
__global__ __launch_bounds__(32) void qkv_gemm(
    const _Float16* __restrict__ xnT, const _Float16* __restrict__ qwh,
    const _Float16* __restrict__ kwh, const _Float16* __restrict__ vwh,
    const float* __restrict__ qb_, const float* __restrict__ kb_,
    const float* __restrict__ vb_, _Float16* __restrict__ qT,
    _Float16* __restrict__ kT, _Float16* __restrict__ vv) {
  const int l0 = blockIdx.x * 16;
  const int o0 = blockIdx.y * 16;
  const int b = blockIdx.z & 3;
  const int which = blockIdx.z >> 2;              // 0=q 1=k 2=v
  const int lane = threadIdx.x & 31;
  const int half = lane >> 4;
  const int ln = lane & 15;

  const _Float16* W = (which == 0) ? qwh : (which == 1) ? kwh : vwh;
  const float* bias = (which == 0) ? qb_ : (which == 1) ? kb_ : vb_;
  const _Float16* xb = xnT + (size_t)b * kL * kC;

  v8f d = {};
#pragma unroll
  for (int kk = 0; kk < 8; ++kk) {
    // A fragment: weight rows o (M), contiguous c chunks at 8*half / 16+8*half
    const _Float16* ap = W + (size_t)(o0 + ln) * kC + kk * 32 + 8 * half;
    v16h a = pack16(*(const v8h*)ap, *(const v8h*)(ap + 16));
    // B fragment: xnT row l (N), contiguous c chunks at 8*half / 16+8*half
    const _Float16* bp = xb + (size_t)(l0 + ln) * kC + kk * 32 + 8 * half;
    v16h bf = pack16(*(const v8h*)bp, *(const v8h*)(bp + 16));
    d = wmma_f16(a, bf, d);
  }

  if (which < 2) {
    _Float16* dst = ((which == 0) ? qT : kT) + (size_t)b * kL * kC;
    v8h pk;
#pragma unroll
    for (int r = 0; r < 8; ++r)
      pk[r] = (_Float16)(d[r] + bias[o0 + 8 * half + r]);
    *(v8h*)(dst + (size_t)(l0 + ln) * kC + o0 + 8 * half) = pk;  // [L,C]
  } else {
    _Float16* dst = vv + (size_t)b * kC * kL;
#pragma unroll
    for (int r = 0; r < 8; ++r)
      dst[(size_t)(o0 + 8 * half + r) * kL + l0 + ln] =
          (_Float16)(d[r] + bias[o0 + 8 * half + r]);           // [C,L]
  }
}

// ---------------------------------------------------------------------------
// 5) Flash attention. 8 waves/block, each wave owns 16 queries.
//    S^T = K*Q^T so softmax stats are per-lane and the S^T D-fragment
//    converts in-lane into the P A-fragment for P*V (no LDS transpose).
//    K/V tiles staged to LDS by TDM, double-buffered on TENSORcnt.
// ---------------------------------------------------------------------------
__global__ __launch_bounds__(256) void flash_attn(
    const _Float16* __restrict__ qT, const _Float16* __restrict__ kT,
    const _Float16* __restrict__ vv, _Float16* __restrict__ aoT) {
  const int b = blockIdx.y;
  const int wave = threadIdx.x >> 5;
  const int lane = threadIdx.x & 31;
  const int half = lane >> 4;
  const int ln = lane & 15;
  const int i0 = (blockIdx.x * 8 + wave) * 16;    // query tile base

  const _Float16* qTb = qT + (size_t)b * kL * kC;
  const _Float16* kTb = kT + (size_t)b * kL * kC;
  const _Float16* vb  = vv + (size_t)b * kC * kL;
  _Float16* aob = aoT + (size_t)b * kL * kC;

  // Double-buffered tiles; rows padded by 8 f16 (16B) against bank conflicts.
  // TDM reproduces the padding via D#.pad_interval/pad_amount.
  __shared__ _Float16 ldsK[2][32 * (kC + 8)];     // [j][c]
  __shared__ _Float16 ldsV[2][kC * 40];           // [c][j]

  // Preload the 8 Q B-fragments (16 queries x 256 c) into VGPRs.
  v16h qf[8];
#pragma unroll
  for (int kk = 0; kk < 8; ++kk) {
    const _Float16* p = qTb + (size_t)(i0 + ln) * kC + kk * 32 + 8 * half;
    qf[kk] = pack16(*(const v8h*)p, *(const v8h*)(p + 16));
  }

  v8f acc[16];                                    // O: 16 queries x 256 c
#pragma unroll
  for (int ct = 0; ct < 16; ++ct) acc[ct] = {};
  float mrow = -1e30f, lrow = 0.f;                // stats for query i0+ln

  // Issue TDM for tile 0 (wave 0 owns the TENSORcnt pipeline).
  if (wave == 0) {
    tdm_load_2d_f16((unsigned)(size_t)&ldsK[0][0], kTb, kC, kL, kC, 32, kC,
                    /*pad_interval=*/6, /*pad_amount=*/3);  // 128dw rows +4dw
    tdm_load_2d_f16((unsigned)(size_t)&ldsV[0][0], vb, kL, kC, 32, kC, kL,
                    /*pad_interval=*/3, /*pad_amount=*/3);  // 16dw rows +4dw
  }

  for (int jt = 0; jt < kL / 32; ++jt) {
    const int cur = jt & 1;
    __syncthreads();   // all waves done reading buffer (jt-1)&1 = !cur
    if (wave == 0) {
      if (jt + 1 < kL / 32) {
        const int j0n = (jt + 1) * 32;
        tdm_load_2d_f16((unsigned)(size_t)&ldsK[cur ^ 1][0],
                        kTb + (size_t)j0n * kC, kC, kL, kC, 32, kC, 6, 3);
        tdm_load_2d_f16((unsigned)(size_t)&ldsV[cur ^ 1][0], vb + j0n, kL, kC,
                        32, kC, kL, 3, 3);
        __builtin_amdgcn_s_wait_tensorcnt(2);  // current tile's 2 ops done
      } else {
        __builtin_amdgcn_s_wait_tensorcnt(0);
      }
    }
    __syncthreads();   // buffer cur visible to all waves

    const _Float16* lk = &ldsK[cur][0];
    const _Float16* lv = &ldsV[cur][0];

    // S^T tiles: rows j (32), cols i (16).  8 K-steps over c.
    v8f st0 = {}, st1 = {};
#pragma unroll
    for (int kk = 0; kk < 8; ++kk) {
      const _Float16* p0 = lk + ln * (kC + 8) + kk * 32 + 8 * half;
      v16h ka0 = pack16(*(const v8h*)p0, *(const v8h*)(p0 + 16));
      const _Float16* p1 = lk + (16 + ln) * (kC + 8) + kk * 32 + 8 * half;
      v16h ka1 = pack16(*(const v8h*)p1, *(const v8h*)(p1 + 16));
      st0 = wmma_f16(ka0, qf[kk], st0);
      st1 = wmma_f16(ka1, qf[kk], st1);
    }

    // Online softmax over j for query i = i0 + ln (per-lane stats).
    float tm = -1e30f;
#pragma unroll
    for (int r = 0; r < 8; ++r) {
      st0[r] *= kScale; st1[r] *= kScale;
      tm = fmaxf(tm, fmaxf(st0[r], st1[r]));
    }
    tm = fmaxf(tm, __shfl_xor(tm, 16, 32));       // merge lane halves
    const float mnew = fmaxf(mrow, tm);
    const float alpha = __expf(mrow - mnew);
    float p0v[8], p1v[8], psum = 0.f;
#pragma unroll
    for (int r = 0; r < 8; ++r) {
      p0v[r] = __expf(st0[r] - mnew);
      p1v[r] = __expf(st1[r] - mnew);
      psum += p0v[r] + p1v[r];
    }
    psum += __shfl_xor(psum, 16, 32);
    lrow = lrow * alpha + psum;
    mrow = mnew;

    // Broadcast alpha to accumulator row owners (O rows live across VGPRs).
    float arow[8];
#pragma unroll
    for (int r = 0; r < 8; ++r) arow[r] = __shfl(alpha, r + 8 * half, 32);

    // P A-fragment built in-lane from the S^T D-fragments (no transpose).
    v16h pa;
#pragma unroll
    for (int r = 0; r < 8; ++r) {
      pa[r] = (_Float16)p0v[r];
      pa[8 + r] = (_Float16)p1v[r];
    }

    // O += P * V over all 16 c-tiles.
#pragma unroll
    for (int ct = 0; ct < 16; ++ct) {
#pragma unroll
      for (int r = 0; r < 8; ++r) acc[ct][r] *= arow[r];
      const _Float16* vp = lv + (ct * 16 + ln) * 40 + 8 * half;
      v16h vbf = pack16(*(const v8h*)vp, *(const v8h*)(vp + 16));
      acc[ct] = wmma_f16(pa, vbf, acc[ct]);
    }
  }

  const float inv = 1.0f / lrow;
  float invr[8];
#pragma unroll
  for (int r = 0; r < 8; ++r) invr[r] = __shfl(inv, r + 8 * half, 32);
#pragma unroll
  for (int ct = 0; ct < 16; ++ct) {
#pragma unroll
    for (int r = 0; r < 8; ++r)
      aob[(size_t)(i0 + r + 8 * half) * kC + ct * 16 + ln] =
          (_Float16)(acc[ct][r] * invr[r]);       // [L,C]
  }
}

// ---------------------------------------------------------------------------
// 6) Output projection + bias + residual (fp32 out).
// ---------------------------------------------------------------------------
__global__ __launch_bounds__(32) void proj_gemm(
    const _Float16* __restrict__ aoT, const _Float16* __restrict__ pwh,
    const float* __restrict__ pb_, const float* __restrict__ x,
    float* __restrict__ out) {
  const int l0 = blockIdx.x * 16;
  const int o0 = blockIdx.y * 16;
  const int b = blockIdx.z;
  const int lane = threadIdx.x & 31;
  const int half = lane >> 4;
  const int ln = lane & 15;
  const _Float16* ab = aoT + (size_t)b * kL * kC;

  v8f d = {};
#pragma unroll
  for (int kk = 0; kk < 8; ++kk) {
    const _Float16* ap = pwh + (size_t)(o0 + ln) * kC + kk * 32 + 8 * half;
    v16h a = pack16(*(const v8h*)ap, *(const v8h*)(ap + 16));
    const _Float16* bp = ab + (size_t)(l0 + ln) * kC + kk * 32 + 8 * half;
    v16h bf = pack16(*(const v8h*)bp, *(const v8h*)(bp + 16));
    d = wmma_f16(a, bf, d);
  }
#pragma unroll
  for (int r = 0; r < 8; ++r) {
    const int o = o0 + 8 * half + r;
    const size_t off = ((size_t)b * kC + o) * kL + l0 + ln;
    out[off] = x[off] + pb_[o] + d[r];
  }
}

// ---------------------------------------------------------------------------
extern "C" void kernel_launch(void* const* d_in, const int* in_sizes, int n_in,
                              void* d_out, int out_size, void* d_ws,
                              size_t ws_size, hipStream_t stream) {
  (void)in_sizes; (void)n_in; (void)out_size; (void)ws_size;
  const float* x   = (const float*)d_in[0];
  const float* gnw = (const float*)d_in[1];
  const float* gnb = (const float*)d_in[2];
  const float* qw  = (const float*)d_in[3];
  const float* qb  = (const float*)d_in[4];
  const float* kw  = (const float*)d_in[5];
  const float* kb  = (const float*)d_in[6];
  const float* vw  = (const float*)d_in[7];
  const float* vb  = (const float*)d_in[8];
  const float* pw  = (const float*)d_in[9];
  const float* pb  = (const float*)d_in[10];
  float* out = (float*)d_out;

  // Workspace layout (~42.5 MB total).
  char* ws = (char*)d_ws;
  size_t off = 0;
  float* mean = (float*)(ws + off); off += 128 * sizeof(float);
  float* rstd = (float*)(ws + off); off += 128 * sizeof(float);
  _Float16* qwh = (_Float16*)(ws + off); off += (size_t)kC * kC * 2;
  _Float16* kwh = (_Float16*)(ws + off); off += (size_t)kC * kC * 2;
  _Float16* vwh = (_Float16*)(ws + off); off += (size_t)kC * kC * 2;
  _Float16* pwh = (_Float16*)(ws + off); off += (size_t)kC * kC * 2;
  const size_t tsz = (size_t)kB * kL * kC * 2;    // 8 MB per f16 tensor
  _Float16* xnT = (_Float16*)(ws + off); off += tsz;
  _Float16* qT  = (_Float16*)(ws + off); off += tsz;
  _Float16* kT  = (_Float16*)(ws + off); off += tsz;
  _Float16* vv  = (_Float16*)(ws + off); off += tsz;
  _Float16* aoT = (_Float16*)(ws + off); off += tsz;

  gn_stats<<<kB * 32, 256, 0, stream>>>(x, mean, rstd);
  gn_apply<<<(kB * kC * kL) / 256, 256, 0, stream>>>(x, gnw, gnb, mean, rstd,
                                                     xnT);
  cvt16<<<(kC * kC) / 256, 256, 0, stream>>>(qw, qwh, kC * kC);
  cvt16<<<(kC * kC) / 256, 256, 0, stream>>>(kw, kwh, kC * kC);
  cvt16<<<(kC * kC) / 256, 256, 0, stream>>>(vw, vwh, kC * kC);
  cvt16<<<(kC * kC) / 256, 256, 0, stream>>>(pw, pwh, kC * kC);
  qkv_gemm<<<dim3(kL / 16, kC / 16, 3 * kB), 32, 0, stream>>>(
      xnT, qwh, kwh, vwh, qb, kb, vb, qT, kT, vv);
  flash_attn<<<dim3(kL / 16 / 8, kB), 256, 0, stream>>>(qT, kT, vv, aoT);
  proj_gemm<<<dim3(kL / 16, kC / 16, kB), 32, 0, stream>>>(aoT, pwh, pb, x,
                                                           out);
}
